// DynamicChannelPruner_7748121002466
// MI455X (gfx1250) — matrix operand order repaired and checked by reading, not compiled, not run on hardware
//
#include <hip/hip_runtime.h>
#include <hip/hip_fp16.h>

typedef __attribute__((ext_vector_type(16))) _Float16 v16h;
typedef __attribute__((ext_vector_type(8)))  float    v8f;
typedef __attribute__((ext_vector_type(4)))  float    f32x4;

// Problem shape
#define BB 32
#define CC 3
#define FF 64
#define HW 4096                 // 64*64
#define MM 96                   // BB*CC
#define NKEEP 32
#define NTOT  ((size_t)BB * CC * FF * HW)   // 25165824 elements
#define NF4   (NTOT / 4)                    // 6291456 float4

__device__ __forceinline__ float sigmoidf_(float x) {
    return 1.0f / (1.0f + expf(-x));
}

// -------------------------------------------------------------------------
// Kernel 1: mean over H*W.  1 wave per (m,f); block = 8 waves; grid = 96*8.
// Coalesced float4 reads; RT temporal hint keeps x_freq resident in L2
// (100.7 MB < 192 MB) so the apply pass re-reads from L2, not HBM.
// -------------------------------------------------------------------------
__global__ void pool_kernel(const float* __restrict__ x, float* __restrict__ xp) {
    int bid  = blockIdx.x;
    int m    = bid >> 3;
    int f    = ((bid & 7) << 3) + (threadIdx.x >> 5);
    int lane = threadIdx.x & 31;

    const f32x4* src = (const f32x4*)(x + ((size_t)m * FF + f) * HW);
    float s = 0.0f;
    #pragma unroll
    for (int it = 0; it < 32; ++it) {
        f32x4 v = src[it * 32 + lane];
        s += v.x + v.y + v.z + v.w;
    }
    #pragma unroll
    for (int off = 16; off; off >>= 1) s += __shfl_xor(s, off, 32);
    if (lane == 0) xp[m * FF + f] = s * (1.0f / 4096.0f);
}

// -------------------------------------------------------------------------
// Kernel 2: the whole "head" in one 768-thread block (24 waves).
// GEMM1: xconv = xp(96x64) @ conv_w^T + conv_b     via v_wmma_f32_16x16x32_f16
// GEMM2: scores = sigmoid(xconv @ fc_w^T + fc_b)   via v_wmma_f32_16x16x32_f16
// 24 waves = 6 Mtiles x 4 Ntiles; K=64 -> 2 WMMAs per tile per GEMM.
// Then row/col means, conv1+BN+sigmoid, convr/convl, outer product,
// blend, stable top-32 mask -> gmask (global ws).
// -------------------------------------------------------------------------
__global__ void __launch_bounds__(768)
head_kernel(const float* __restrict__ xp,
            const float* __restrict__ conv_w, const float* __restrict__ conv_b,
            const float* __restrict__ fc_w,   const float* __restrict__ fc_b,
            const float* __restrict__ conv1_w, const float* __restrict__ conv1_b,
            const float* __restrict__ convr_w, const float* __restrict__ convr_b,
            const float* __restrict__ convl_w, const float* __restrict__ convl_b,
            const float* __restrict__ bn_gamma, const float* __restrict__ bn_beta,
            const float* __restrict__ bn_mean,  const float* __restrict__ bn_var,
            const float* __restrict__ a_scalar,
            float* __restrict__ gmask) {
    __shared__ _Float16 s_xp[MM * FF];   // 12288 B  pooled input (A of GEMM1)
    __shared__ _Float16 s_w [FF * FF];   //  8192 B  staged weight (conv_w, then fc_w)
    __shared__ _Float16 s_xc[MM * FF];   // 12288 B  xconv (A of GEMM2)
    __shared__ float    s_sc[MM * FF];   // 24576 B  scores -> final scores
    __shared__ float    s_ar[MM * 8];    //  3072 B
    __shared__ float    s_al[MM * 8];    //  3072 B   (total 63488 B)

    const int tid  = threadIdx.x;
    const int lane = tid & 31;
    const int wave = tid >> 5;           // 0..23
    const int hi   = lane >> 4;          // lane half
    const int lo   = lane & 15;
    const int Mb   = (wave % 6) * 16;
    const int Nb   = (wave / 6) * 16;

    // Stage pooled activations + conv_w as f16
    for (int i = tid; i < MM * FF; i += 768) s_xp[i] = (_Float16)xp[i];
    for (int i = tid; i < FF * FF; i += 768) s_w[i]  = (_Float16)conv_w[i];
    __syncthreads();

    // ---- GEMM1: D = A(16x32 f16) x B(32x16 f16) + C, K=64 in 2 steps ----
    // A layout (ISA 7.12.2, 16-bit A 16x32): lane lo = M row, element j:
    //   r=j/2, K = 2r + (r>=4?8:0) + (j&1) + 8*hi
    // B layout (16-bit B 32x16): lane lo = N col, element j: K = j + 8*hi + (j>=8?8:0)
    v8f acc = {};
    #pragma unroll
    for (int k0 = 0; k0 < 64; k0 += 32) {
        v16h afr, bfr;
        #pragma unroll
        for (int j = 0; j < 16; ++j) {
            int r   = j >> 1;
            int kkA = 2 * r + ((r >= 4) ? 8 : 0) + (j & 1) + 8 * hi;
            int kkB = j + 8 * hi + ((j >= 8) ? 8 : 0);
            afr[j] = s_xp[(Mb + lo) * FF + k0 + kkA];
            bfr[j] = s_w [(Nb + lo) * FF + k0 + kkB];   // B[k][n] = conv_w[n][k]
        }
        acc = __builtin_amdgcn_wmma_f32_16x16x32_f16(
                  false, afr, false, bfr, (short)0, acc, false, false);
    }
    {   // D layout: VGPR i -> M = Mb + i + 8*hi, N = Nb + lo
        float bias = conv_b[Nb + lo];
        #pragma unroll
        for (int i = 0; i < 8; ++i)
            s_xc[(Mb + i + 8 * hi) * FF + (Nb + lo)] = (_Float16)(acc[i] + bias);
    }
    __syncthreads();

    // restage fc_w
    for (int i = tid; i < FF * FF; i += 768) s_w[i] = (_Float16)fc_w[i];
    __syncthreads();

    // ---- GEMM2: scores = sigmoid(xconv @ fc_w^T + fc_b) ----
    v8f acc2 = {};
    #pragma unroll
    for (int k0 = 0; k0 < 64; k0 += 32) {
        v16h afr, bfr;
        #pragma unroll
        for (int j = 0; j < 16; ++j) {
            int r   = j >> 1;
            int kkA = 2 * r + ((r >= 4) ? 8 : 0) + (j & 1) + 8 * hi;
            int kkB = j + 8 * hi + ((j >= 8) ? 8 : 0);
            afr[j] = s_xc[(Mb + lo) * FF + k0 + kkA];
            bfr[j] = s_w [(Nb + lo) * FF + k0 + kkB];
        }
        acc2 = __builtin_amdgcn_wmma_f32_16x16x32_f16(
                   false, afr, false, bfr, (short)0, acc2, false, false);
    }
    {
        float bias = fc_b[Nb + lo];
        #pragma unroll
        for (int i = 0; i < 8; ++i)
            s_sc[(Mb + i + 8 * hi) * FF + (Nb + lo)] = sigmoidf_(acc2[i] + bias);
    }
    __syncthreads();

    // ---- tiny spatial-attention chain: one thread per batch b ----
    if (tid < BB) {
        float rowm[CC][8], colm[CC][8];
        for (int c = 0; c < CC; ++c) {
            const _Float16* row = &s_xp[(tid * CC + c) * FF];
            for (int i = 0; i < 8; ++i) { rowm[c][i] = 0.f; colm[c][i] = 0.f; }
            for (int i = 0; i < 8; ++i)
                for (int j = 0; j < 8; ++j) {
                    float v = (float)row[i * 8 + j];
                    rowm[c][i] += v; colm[c][j] += v;
                }
            for (int i = 0; i < 8; ++i) { rowm[c][i] *= 0.125f; colm[c][i] *= 0.125f; }
        }
        // conv1 (channel mix) + BN + sigmoid ; row-branch -> xr, col-branch -> xl
        float xr[CC][8], xl[CC][8];
        for (int d = 0; d < CC; ++d) {
            float inv = rsqrtf(bn_var[d] + 1e-5f);
            float g = bn_gamma[d], bt = bn_beta[d], mu = bn_mean[d], cb = conv1_b[d];
            for (int i = 0; i < 8; ++i) {
                float y0 = cb, y1 = cb;
                for (int c = 0; c < CC; ++c) {
                    y0 += rowm[c][i] * conv1_w[d * CC + c];
                    y1 += colm[c][i] * conv1_w[d * CC + c];
                }
                xr[d][i] = sigmoidf_((y0 - mu) * inv * g + bt);
                xl[d][i] = sigmoidf_((y1 - mu) * inv * g + bt);
            }
        }
        // convr / convl channel mixes + sigmoid
        for (int d = 0; d < CC; ++d) {
            for (int i = 0; i < 8; ++i) {
                float ar = convr_b[d], al = convl_b[d];
                for (int c = 0; c < CC; ++c) {
                    ar += xr[c][i] * convr_w[d * CC + c];
                    al += xl[c][i] * convl_w[d * CC + c];
                }
                s_ar[(tid * CC + d) * 8 + i] = sigmoidf_(ar);
                s_al[(tid * CC + d) * 8 + i] = sigmoidf_(al);
            }
        }
    }
    __syncthreads();

    // ---- blend: final = a * (a_r ⊗ a_l) + (1-a) * scores ----
    float av = a_scalar[0];
    for (int i = tid; i < MM * FF; i += 768) {
        int m = i >> 6, f = i & 63;
        float att = s_ar[m * 8 + (f >> 3)] * s_al[m * 8 + (f & 7)];
        s_sc[i] = av * att + (1.0f - av) * s_sc[i];
    }
    __syncthreads();

    // ---- stable top-32 mask per (b,c), matching jax.lax.top_k tie-breaking ----
    if (tid < MM) {
        const float* row = &s_sc[tid * FF];
        for (int f = 0; f < FF; ++f) {
            float v = row[f];
            int rank = 0;
            for (int f2 = 0; f2 < FF; ++f2) {
                float v2 = row[f2];
                rank += (v2 > v) || (v2 == v && f2 < f);
            }
            gmask[tid * FF + f] = (rank < NKEEP) ? 1.0f : 0.0f;
        }
    }
}

// -------------------------------------------------------------------------
// Kernel 3: x_pruned = x_freq * mask ; x_pruned_2k = 0.
// Grid-stride float4; NT stores for the 201 MB write-once output; prefetch
// one stride ahead (global_prefetch_b8). x_freq read hits L2 from kernel 1.
// -------------------------------------------------------------------------
__global__ void apply_kernel(const float* __restrict__ x,
                             const float* __restrict__ gmask,
                             float* __restrict__ out,
                             float* __restrict__ outz) {
    const f32x4* src  = (const f32x4*)x;
    f32x4*       dst  = (f32x4*)out;
    f32x4*       dstz = (f32x4*)outz;
    const size_t stride = (size_t)gridDim.x * blockDim.x;
    const f32x4 zero = (f32x4)(0.0f);

    for (size_t i = (size_t)blockIdx.x * blockDim.x + threadIdx.x; i < NF4; i += stride) {
        __builtin_prefetch(&src[i + stride], 0, 1);
        float mv = gmask[(int)(i >> 10)];        // 1024 float4 per (b,c,f) slice
        f32x4 v = src[i];
        v *= mv;
        __builtin_nontemporal_store(v,    &dst[i]);
        __builtin_nontemporal_store(zero, &dstz[i]);
    }
}

// -------------------------------------------------------------------------
extern "C" void kernel_launch(void* const* d_in, const int* in_sizes, int n_in,
                              void* d_out, int out_size, void* d_ws, size_t ws_size,
                              hipStream_t stream) {
    const float* x_freq  = (const float*)d_in[0];
    const float* conv_w  = (const float*)d_in[1];
    const float* conv_b  = (const float*)d_in[2];
    const float* fc_w    = (const float*)d_in[3];
    const float* fc_b    = (const float*)d_in[4];
    const float* conv1_w = (const float*)d_in[5];
    const float* conv1_b = (const float*)d_in[6];
    const float* convr_w = (const float*)d_in[7];
    const float* convr_b = (const float*)d_in[8];
    const float* convl_w = (const float*)d_in[9];
    const float* convl_b = (const float*)d_in[10];
    const float* bn_g    = (const float*)d_in[11];
    const float* bn_b    = (const float*)d_in[12];
    const float* bn_m    = (const float*)d_in[13];
    const float* bn_v    = (const float*)d_in[14];
    const float* a_sc    = (const float*)d_in[15];

    float* xp    = (float*)d_ws;            // 96*64 f32
    float* gmask = xp + MM * FF;            // 96*64 f32

    float* out  = (float*)d_out;            // x_pruned
    float* out2 = out + NTOT;               // x_pruned_2k (zeros)

    pool_kernel<<<MM * 8, 256, 0, stream>>>(x_freq, xp);
    head_kernel<<<1, 768, 0, stream>>>(xp, conv_w, conv_b, fc_w, fc_b,
                                       conv1_w, conv1_b, convr_w, convr_b,
                                       convl_w, convl_b, bn_g, bn_b, bn_m, bn_v,
                                       a_sc, gmask);
    apply_kernel<<<6144, 256, 0, stream>>>(x_freq, gmask, out, out2);
}
